// StructLinearLSTM_31473520345394
// MI455X (gfx1250) — matrix operand order repaired and checked by reading, not compiled, not run on hardware
//
#include <hip/hip_runtime.h>
#include <hip/hip_bf16.h>
#include <stdint.h>

// ---------------- fixed problem dims (from reference) ----------------
#define BB    128
#define NSAMP 4
#define BN    512          // B*NS
#define HH    512
#define EE    512
#define VV    512
#define TT    100
#define MM    256
#define LL    2

typedef __bf16 bf16_t;
typedef bf16_t v16bf __attribute__((ext_vector_type(16)));
typedef float  v8f   __attribute__((ext_vector_type(8)));

__device__ __forceinline__ uint16_t f2bf(float f) {
  uint32_t u = __float_as_uint(f);
  uint32_t r = u + 0x7FFFu + ((u >> 16) & 1u);
  return (uint16_t)(r >> 16);
}

__device__ __forceinline__ float sigm(float x) { return 1.0f / (1.0f + __expf(-x)); }

__device__ __forceinline__ float urand(uint32_t a, uint32_t b, uint32_t c) {
  uint32_t x = a * 0x9E3779B9u ^ b * 0x85EBCA6Bu ^ c * 0xC2B2AE35u ^ 0x6C078965u;
  x ^= x >> 16; x *= 0x7FEB352Du; x ^= x >> 15; x *= 0x846CA68Bu; x ^= x >> 16;
  return (float)((x >> 8) + 1u) * (1.0f / 16777217.0f);
}

// A fragment: lane holds its K-pairs at [base..base+7] and [base+16..base+23]
// (16-bit A 16x32 layout; base already includes hi*8). Two 16B chunks.
__device__ __forceinline__ v16bf load_afrag(const uint16_t* ab) {
  v16bf a;
  __builtin_memcpy(&a, ab, 16);
  __builtin_memcpy(reinterpret_cast<char*>(&a) + 16, ab + 16, 16);
  return a;
}

__device__ __forceinline__ v8f wmma_bf16_acc(v16bf a, v16bf b, v8f c) {
  return __builtin_amdgcn_wmma_f32_16x16x32_bf16(false, a, false, b, (short)0, c, false, false);
}

// ---- weight pack: W[N,K] f32 -> fragment-major bf16 B tiles ----------
// Tile (kt, nt) = 32x16 B fragment; lane L's 16 bf16 values contiguous:
//   elem = ((kt*(N/16) + nt)*32 + lane)*16 + kk
// where lane = hi*16 + (n%16), hi = ((k%32)/8)&1, kk = (k%8) + 8*((k%32)/16).
__global__ __launch_bounds__(256) void k_pack(const float* __restrict__ W,
                                              uint16_t* __restrict__ out,
                                              int N, int K) {
  int id = blockIdx.x * 256 + threadIdx.x;
  if (id >= N * K) return;
  int n = id / K, k = id - n * K;
  int kt = k >> 5, rem = k & 31;
  int hi = (rem >> 3) & 1, kk = (rem & 7) | ((rem >> 4) << 3);
  int lane = (hi << 4) | (n & 15);
  size_t tile = (size_t)kt * (N >> 4) + (n >> 4);
  out[(tile * 32 + lane) * 16 + kk] = f2bf(W[id]);
}

__global__ __launch_bounds__(256) void k_cvt(const float* __restrict__ src,
                                             uint16_t* __restrict__ dst, int n) {
  int id = blockIdx.x * 256 + threadIdx.x;
  if (id < n) dst[id] = f2bf(src[id]);
}

__global__ __launch_bounds__(256) void k_yrep(const float* __restrict__ y,
                                              uint16_t* __restrict__ o) {
  int id = blockIdx.x * 256 + threadIdx.x;        // BN*HH
  int bn = id >> 9, j = id & (HH - 1);
  o[id] = f2bf(y[(size_t)(bn >> 2) * HH + j]);    // sample-minor repeat
}

__global__ __launch_bounds__(256) void k_bias2(const float* __restrict__ a,
                                               const float* __restrict__ b,
                                               float* __restrict__ o, int n) {
  int id = blockIdx.x * 256 + threadIdx.x;
  if (id < n) o[id] = a[id] + b[id];
}

__global__ __launch_bounds__(256) void k_init(const uint16_t* __restrict__ embBf,
                                              uint16_t* __restrict__ xbf,
                                              float* __restrict__ zlp,
                                              float* __restrict__ ent) {
  int id = blockIdx.x * 256 + threadIdx.x;
  if (id < BN * EE) xbf[id] = embBf[(size_t)(VV - 1) * EE + (id & (EE - 1))];
  if (id < BN) zlp[id] = 0.0f;
  if (id == 0) *ent = 0.0f;
}

// ---------------- WMMA GEMM: (MT*16) x 32 output tile per wave --------
// C[M,N] = A1[M,K1]*B1 + (A2 ? A2[M,K2]*B2 : 0) + bias
// A*: bf16 row-major (lda*); B*: fragment-major packed tiles (see k_pack).
// 8 waves / 256-thread block. MT=2: 32x32 tile, 4 WMMAs per K-step off
// 8 b128 loads (2 loads/WMMA). MT=1: 16x32 tile (for small-N occupancy).
template <int MT>
__global__ __launch_bounds__(256) void k_gemm(
    const uint16_t* __restrict__ A1, int lda1, const uint16_t* __restrict__ B1, int K1,
    const uint16_t* __restrict__ A2, int lda2, const uint16_t* __restrict__ B2, int K2,
    const float* __restrict__ bias, float* __restrict__ C, uint16_t* __restrict__ Cbf,
    int Mrows, int N) {
  int ntilesN = N >> 4;                      // 16-wide B tiles per K-slice
  int ntiles2 = N >> 5;                      // 32-wide wave tiles
  int tiles   = ((Mrows >> 4) / MT) * ntiles2;
  int tile    = blockIdx.x * 8 + (threadIdx.x >> 5);
  if (tile >= tiles) return;                 // uniform per wave: EXEC all-1s
  int lane = threadIdx.x & 31;
  int ln   = lane & 15;
  int hi   = lane >> 4;
  int mt   = (tile / ntiles2) * MT;
  int nt   = (tile - (tile / ntiles2) * ntiles2) << 1;
  int row0 = (mt << 4) + ln;

  size_t bstride = (size_t)ntilesN << 9;     // elems per K-slice
  v8f acc[MT][2];
#pragma unroll
  for (int i = 0; i < MT; ++i) { acc[i][0] = (v8f){}; acc[i][1] = (v8f){}; }
  {
    const uint16_t* ab = A1 + (size_t)row0 * lda1 + hi * 8;
    const uint16_t* bb = B1 + (size_t)((nt * 32) + lane) * 16;
    for (int k0 = 0; k0 < K1; k0 += 32) {
      v16bf b0 = *reinterpret_cast<const v16bf*>(bb);
      v16bf b1 = *reinterpret_cast<const v16bf*>(bb + 512);
#pragma unroll
      for (int i = 0; i < MT; ++i) {
        v16bf a = load_afrag(ab + (size_t)i * 16 * lda1);
        acc[i][0] = wmma_bf16_acc(a, b0, acc[i][0]);
        acc[i][1] = wmma_bf16_acc(a, b1, acc[i][1]);
      }
      ab += 32;
      bb += bstride;
    }
  }
  if (A2) {
    const uint16_t* ab = A2 + (size_t)row0 * lda2 + hi * 8;
    const uint16_t* bb = B2 + (size_t)((nt * 32) + lane) * 16;
    for (int k0 = 0; k0 < K2; k0 += 32) {
      v16bf b0 = *reinterpret_cast<const v16bf*>(bb);
      v16bf b1 = *reinterpret_cast<const v16bf*>(bb + 512);
#pragma unroll
      for (int i = 0; i < MT; ++i) {
        v16bf a = load_afrag(ab + (size_t)i * 16 * lda2);
        acc[i][0] = wmma_bf16_acc(a, b0, acc[i][0]);
        acc[i][1] = wmma_bf16_acc(a, b1, acc[i][1]);
      }
      ab += 32;
      bb += bstride;
    }
  }
  int n0 = (nt << 4) + ln;
  int n1 = n0 + 16;
  float bv0 = bias ? bias[n0] : 0.0f;
  float bv1 = bias ? bias[n1] : 0.0f;
#pragma unroll
  for (int i = 0; i < MT; ++i) {
#pragma unroll
    for (int v = 0; v < 8; ++v) {
      int m = ((mt + i) << 4) + v + (hi << 3);   // C/D layout: lanes16-31 -> M+8
      float v0 = acc[i][0][v] + bv0;
      float v1 = acc[i][1][v] + bv1;
      C[(size_t)m * N + n0] = v0;
      C[(size_t)m * N + n1] = v1;
      if (Cbf) {
        Cbf[(size_t)m * N + n0] = f2bf(v0);
        Cbf[(size_t)m * N + n1] = f2bf(v1);
      }
    }
  }
}

// ---------------- LSTM gates (PyTorch order i,f,g,o) ------------------
__global__ __launch_bounds__(256) void k_gates(const float* __restrict__ g,
                                               float* __restrict__ cbuf,
                                               uint16_t* __restrict__ hbf,
                                               uint16_t* __restrict__ catbf,
                                               int layer) {
  int id = blockIdx.x * 256 + threadIdx.x;   // BN*HH threads
  int bn = id >> 9, j = id & (HH - 1);
  const float* gr = g + (size_t)bn * (4 * HH);
  float i_ = gr[j], f_ = gr[HH + j], gg = gr[2 * HH + j], o_ = gr[3 * HH + j];
  float* cp = cbuf + (size_t)bn * (LL * HH) + layer * HH + j;
  float cn = sigm(f_) * (*cp) + sigm(i_) * tanhf(gg);
  float hn = sigm(o_) * tanhf(cn);
  *cp = cn;
  uint16_t hb = f2bf(hn);
  hbf[(size_t)bn * (LL * HH) + layer * HH + j] = hb;
  if (catbf) catbf[(size_t)bn * (2 * HH) + j] = hb;  // first half of [out,ctx]
}

// ---------------- attention scores: q . kv + mask ---------------------
__global__ __launch_bounds__(256) void k_scores(const float* __restrict__ q,
                                                const float* __restrict__ kv,
                                                const unsigned char* __restrict__ kvmask,
                                                float* __restrict__ scores) {
  int id = blockIdx.x * 256 + threadIdx.x;   // BN*MM threads
  int bn = id >> 8, m = id & (MM - 1);
  int b  = bn >> 2;                          // sample-minor repeat
  const float4* qr = reinterpret_cast<const float4*>(q + (size_t)bn * HH);
  const float4* kr = reinterpret_cast<const float4*>(kv + ((size_t)b * MM + m) * HH);
  float acc = 0.0f;
#pragma unroll 4
  for (int k = 0; k < HH / 4; ++k) {
    float4 a = qr[k], c = kr[k];
    acc += a.x * c.x + a.y * c.y + a.z * c.z + a.w * c.w;
  }
  acc += kvmask[b * MM + m] ? 0.0f : -1e9f;
  scores[id] = acc;
}

// ---------------- softmax over M + context, write bf16 into cat -------
__global__ __launch_bounds__(256) void k_attn(const float* __restrict__ scores,
                                              const float* __restrict__ kv,
                                              uint16_t* __restrict__ catbf) {
  __shared__ float red[256];
  __shared__ float p[256];
  int bn = blockIdx.x, tid = threadIdx.x, b = bn >> 2;
  float s = scores[(size_t)bn * MM + tid];
  red[tid] = s; __syncthreads();
  for (int o = 128; o; o >>= 1) { if (tid < o) red[tid] = fmaxf(red[tid], red[tid + o]); __syncthreads(); }
  float mx = red[0]; __syncthreads();
  float e = __expf(s - mx);
  red[tid] = e; __syncthreads();
  for (int o = 128; o; o >>= 1) { if (tid < o) red[tid] += red[tid + o]; __syncthreads(); }
  float S = red[0]; __syncthreads();
  p[tid] = e / S; __syncthreads();
  const float* kb = kv + (size_t)b * MM * HH;
  for (int d = tid; d < HH; d += 256) {
    float acc = 0.0f;
    for (int m = 0; m < MM; ++m) acc += p[m] * kb[(size_t)m * HH + d];
    catbf[(size_t)bn * (2 * HH) + HH + d] = f2bf(acc);
  }
}

// ---------------- log-softmax + Gumbel sample + outputs ---------------
__global__ __launch_bounds__(256) void k_sample(const float* __restrict__ logits,
                                                const float* __restrict__ emb,
                                                const uint16_t* __restrict__ embBf,
                                                const float* __restrict__ zmask,
                                                float* __restrict__ zids,
                                                float* __restrict__ zstates,
                                                float* __restrict__ zlp,
                                                float* __restrict__ stepw,
                                                float* __restrict__ entacc,
                                                uint16_t* __restrict__ xbf, int t) {
  __shared__ float red[256];
  __shared__ float red2[256];
  __shared__ int   ridx[256];
  int bn = blockIdx.x, tid = threadIdx.x, b = bn >> 2;
  const float* lr = logits + (size_t)bn * VV;
  float l0 = lr[tid], l1 = lr[tid + 256];
  red[tid] = fmaxf(l0, l1); __syncthreads();
  for (int o = 128; o; o >>= 1) { if (tid < o) red[tid] = fmaxf(red[tid], red[tid + o]); __syncthreads(); }
  float mx = red[0]; __syncthreads();
  float e0 = __expf(l0 - mx), e1 = __expf(l1 - mx);
  red[tid] = e0 + e1; red2[tid] = e0 * l0 + e1 * l1; __syncthreads();
  for (int o = 128; o; o >>= 1) {
    if (tid < o) { red[tid] += red[tid + o]; red2[tid] += red2[tid + o]; }
    __syncthreads();
  }
  float S = red[0], SL = red2[0]; __syncthreads();
  // Gumbel-argmax categorical sample (hash RNG keyed by (t, bn, v))
  float g0 = l0 - __logf(-__logf(urand((uint32_t)t, (uint32_t)bn, (uint32_t)tid)));
  float g1 = l1 - __logf(-__logf(urand((uint32_t)t, (uint32_t)bn, (uint32_t)(tid + 256))));
  float bv; int bi;
  if (g0 >= g1) { bv = g0; bi = tid; } else { bv = g1; bi = tid + 256; }
  red[tid] = bv; ridx[tid] = bi; __syncthreads();
  for (int o = 128; o; o >>= 1) {
    if (tid < o && red[tid + o] > red[tid]) { red[tid] = red[tid + o]; ridx[tid] = ridx[tid + o]; }
    __syncthreads();
  }
  int idx = ridx[0];
  if (tid == 0) {
    float logZ = mx + __logf(S);
    float slp  = lr[idx] - logZ;
    float ent  = logZ - SL / S;
    zids[(size_t)bn * TT + t]  = (float)idx;
    stepw[(size_t)bn * TT + t] = slp;
    zlp[bn] += slp * zmask[b * TT + t];
    atomicAdd(entacc, ent * (1.0f / ((float)BN * (float)TT)));
  }
  const float*    er  = emb   + (size_t)idx * EE;
  const uint16_t* ebr = embBf + (size_t)idx * EE;
  float* zs = zstates + ((size_t)bn * TT + t) * EE;
  for (int d = tid; d < EE; d += 256) {
    zs[d] = er[d];
    xbf[(size_t)bn * EE + d] = ebr[d];
  }
}

// ---------------- host driver ----------------------------------------
extern "C" void kernel_launch(void* const* d_in, const int* in_sizes, int n_in,
                              void* d_out, int out_size, void* d_ws, size_t ws_size,
                              hipStream_t stream) {
  (void)in_sizes; (void)n_in; (void)out_size; (void)ws_size;
  const float* y        = (const float*)d_in[0];
  const float* kv       = (const float*)d_in[1];
  const unsigned char* kvmask = (const unsigned char*)d_in[2];
  const float* zmask    = (const float*)d_in[3];
  const float* emb      = (const float*)d_in[4];
  const float* W_h0     = (const float*)d_in[5];
  const float* b_h0     = (const float*)d_in[6];
  const float* W_c0     = (const float*)d_in[7];
  const float* b_c0     = (const float*)d_in[8];
  const float* W_ih     = (const float*)d_in[9];
  const float* W_hh     = (const float*)d_in[10];
  const float* b_ih     = (const float*)d_in[11];
  const float* b_hh     = (const float*)d_in[12];
  const float* W_attn   = (const float*)d_in[13];
  const float* W_cproj  = (const float*)d_in[14];
  const float* b_cproj  = (const float*)d_in[15];
  const float* W_out    = (const float*)d_in[16];
  const float* b_out    = (const float*)d_in[17];

  // d_out layout: z_ids | z_states | z_lp | stepwise | ent
  float* o_zids    = (float*)d_out;
  float* o_zstates = o_zids + (size_t)BN * TT;
  float* o_zlp     = o_zstates + (size_t)BN * TT * EE;
  float* o_stepw   = o_zlp + BN;
  float* o_ent     = o_stepw + (size_t)BN * TT;

  // workspace carve-up (bytes, 256-aligned)
  char* ws = (char*)d_ws;
  size_t off = 0;
  auto take = [&](size_t bytes) { char* p = ws + off; off = (off + bytes + 255) & ~(size_t)255; return p; };
  uint16_t* Wh0P   = (uint16_t*)take((size_t)2 * 1024 * HH);
  uint16_t* Wc0P   = (uint16_t*)take((size_t)2 * 1024 * HH);
  uint16_t* WihP0  = (uint16_t*)take((size_t)2 * 2048 * EE);
  uint16_t* WihP1  = (uint16_t*)take((size_t)2 * 2048 * EE);
  uint16_t* WhhP0  = (uint16_t*)take((size_t)2 * 2048 * HH);
  uint16_t* WhhP1  = (uint16_t*)take((size_t)2 * 2048 * HH);
  uint16_t* WattnP = (uint16_t*)take((size_t)2 * HH * HH);
  uint16_t* WcprP  = (uint16_t*)take((size_t)2 * HH * (2 * HH));
  uint16_t* WoutP  = (uint16_t*)take((size_t)2 * VV * HH);
  uint16_t* embBf  = (uint16_t*)take((size_t)2 * VV * EE);
  uint16_t* yBf    = (uint16_t*)take((size_t)2 * BN * HH);
  uint16_t* hBf    = (uint16_t*)take((size_t)2 * BN * LL * HH);       // [BN][L*H]
  uint16_t* xBf    = (uint16_t*)take((size_t)2 * BN * EE);
  uint16_t* catBf  = (uint16_t*)take((size_t)2 * BN * 2 * HH);
  uint16_t* out2Bf = (uint16_t*)take((size_t)2 * BN * HH);
  float* cbuf   = (float*)take((size_t)4 * BN * LL * HH);
  float* gbuf   = (float*)take((size_t)4 * BN * 4 * HH);              // also h-init scratch
  float* qbuf   = (float*)take((size_t)4 * BN * HH);
  float* sbuf   = (float*)take((size_t)4 * BN * MM);
  float* out2   = (float*)take((size_t)4 * BN * HH);
  float* logit  = (float*)take((size_t)4 * BN * VV);
  float* bcomb  = (float*)take((size_t)4 * LL * 4 * HH);

  const int TPB = 256;
  auto blk = [](int n) { return (n + 255) / 256; };
  // wave-tiles per GEMM grid: ((M/16)/MT)*(N/32)/8 blocks
  auto gblk = [](int M, int N, int MT) { return (((M >> 4) / MT) * (N >> 5) + 7) / 8; };

  // ---- one-time per call: pack weights to fragment-major bf16 ---------
  k_pack<<<blk(1024 * HH), TPB, 0, stream>>>(W_h0, Wh0P, 1024, HH);
  k_pack<<<blk(1024 * HH), TPB, 0, stream>>>(W_c0, Wc0P, 1024, HH);
  k_pack<<<blk(2048 * EE), TPB, 0, stream>>>(W_ih,                     WihP0, 2048, EE);
  k_pack<<<blk(2048 * EE), TPB, 0, stream>>>(W_ih + (size_t)2048 * EE, WihP1, 2048, EE);
  k_pack<<<blk(2048 * HH), TPB, 0, stream>>>(W_hh,                     WhhP0, 2048, HH);
  k_pack<<<blk(2048 * HH), TPB, 0, stream>>>(W_hh + (size_t)2048 * HH, WhhP1, 2048, HH);
  k_pack<<<blk(HH * HH), TPB, 0, stream>>>(W_attn, WattnP, HH, HH);
  k_pack<<<blk(HH * 2 * HH), TPB, 0, stream>>>(W_cproj, WcprP, HH, 2 * HH);
  k_pack<<<blk(VV * HH), TPB, 0, stream>>>(W_out, WoutP, VV, HH);
  k_cvt<<<blk(VV * EE), TPB, 0, stream>>>(emb, embBf, VV * EE);
  k_yrep<<<blk(BN * HH), TPB, 0, stream>>>(y, yBf);
  k_bias2<<<blk(LL * 4 * HH), TPB, 0, stream>>>(b_ih, b_hh, bcomb, LL * 4 * HH);
  k_init<<<blk(BN * EE), TPB, 0, stream>>>(embBf, xBf, o_zlp, o_ent);

  // ---- initial state: h = yRep @ Wh0^T + b_h0 ; c likewise ------------
  k_gemm<2><<<gblk(BN, 1024, 2), TPB, 0, stream>>>(
      yBf, HH, Wh0P, HH, nullptr, 0, nullptr, 0, b_h0, gbuf, hBf, BN, 1024);
  k_gemm<2><<<gblk(BN, 1024, 2), TPB, 0, stream>>>(
      yBf, HH, Wc0P, HH, nullptr, 0, nullptr, 0, b_c0, cbuf, nullptr, BN, 1024);

  // ---- T sequential decode steps --------------------------------------
  for (int t = 0; t < TT; ++t) {
    // layer 0: g = x*Wih0 + h0*Whh0 + b
    k_gemm<2><<<gblk(BN, 2048, 2), TPB, 0, stream>>>(
        xBf, EE, WihP0, EE, hBf, LL * HH, WhhP0, HH, bcomb, gbuf, nullptr, BN, 2048);
    k_gates<<<blk(BN * HH), TPB, 0, stream>>>(gbuf, cbuf, hBf, nullptr, 0);
    // layer 1: g = h0*Wih1 + h1*Whh1 + b
    k_gemm<2><<<gblk(BN, 2048, 2), TPB, 0, stream>>>(
        hBf, LL * HH, WihP1, HH, hBf + HH, LL * HH, WhhP1, HH, bcomb + 4 * HH,
        gbuf, nullptr, BN, 2048);
    k_gates<<<blk(BN * HH), TPB, 0, stream>>>(gbuf, cbuf, hBf, catBf, 1);
    // q = out @ Wattn^T
    k_gemm<1><<<gblk(BN, HH, 1), TPB, 0, stream>>>(
        hBf + HH, LL * HH, WattnP, HH, nullptr, 0, nullptr, 0, nullptr, qbuf, nullptr, BN, HH);
    k_scores<<<blk(BN * MM), TPB, 0, stream>>>(qbuf, kv, kvmask, sbuf);
    k_attn<<<BN, TPB, 0, stream>>>(sbuf, kv, catBf);
    // out2 = [out,ctx] @ Wcproj^T + b
    k_gemm<1><<<gblk(BN, HH, 1), TPB, 0, stream>>>(
        catBf, 2 * HH, WcprP, 2 * HH, nullptr, 0, nullptr, 0, b_cproj, out2, out2Bf, BN, HH);
    // logits = out2 @ Wout^T + b
    k_gemm<1><<<gblk(BN, VV, 1), TPB, 0, stream>>>(
        out2Bf, HH, WoutP, HH, nullptr, 0, nullptr, 0, b_out, logit, nullptr, BN, VV);
    k_sample<<<BN, TPB, 0, stream>>>(logit, emb, embBf, zmask, o_zids, o_zstates,
                                     o_zlp, o_stepw, o_ent, xBf, t);
  }
}
